// PointNetpp_28200755265730
// MI455X (gfx1250) — compile-verified, matrix operand order checked by reading, not run on hardware
//
#include <hip/hip_runtime.h>

typedef __attribute__((ext_vector_type(16))) _Float16 v16h;
typedef __attribute__((ext_vector_type(8)))  float    v8f;

#define BN_EPS 1e-5f

// ---------------------------------------------------------------------------
// Tensor Data Mover staging (gfx1250). D# built per CDNA5 ISA ch.8.
// ---------------------------------------------------------------------------
#if __has_builtin(__builtin_amdgcn_tensor_load_to_lds) && \
    __has_builtin(__builtin_amdgcn_s_wait_tensorcnt)
#define USE_TDM 1
typedef __attribute__((ext_vector_type(4))) unsigned int u32x4;
typedef __attribute__((ext_vector_type(4))) int          i32x4;
typedef __attribute__((ext_vector_type(8))) int          i32x8;

// Copy nelem 2-byte elements (contiguous) from global to LDS via TDM.
__device__ __forceinline__ void tdm_copy_to_lds(const void* gsrc, void* ldst,
                                                unsigned nelem) {
  const unsigned lds_off = (unsigned)(unsigned long long)ldst;  // LDS byte offset
  const unsigned long long ga = (unsigned long long)gsrc;
  u32x4 g0;
  g0[0] = 1u;                                    // count=1 (valid user D#)
  g0[1] = lds_off;                               // lds_addr
  g0[2] = (unsigned)(ga & 0xFFFFFFFFu);          // global_addr[31:0]
  g0[3] = (unsigned)((ga >> 32) & 0x1FFFFFFu)    // global_addr[56:32]
          | (2u << 30);                          // type=2 ("image")
  i32x8 g1;
  const unsigned td0 = nelem;                    // one row of nelem elements
  g1[0] = (int)(1u << 16);                       // wg_mask=0, data_size=1 (2B)
  g1[1] = (int)((td0 & 0xFFFFu) << 16);          // tensor_dim0[15:0] @ [63:48]
  g1[2] = (int)(((td0 >> 16) & 0xFFFFu)          // tensor_dim0[31:16]
                | (1u << 16));                   // tensor_dim1 = 1
  g1[3] = (int)((td0 & 0xFFFFu) << 16);          // tile_dim0 @ [127:112]
  g1[4] = 1;                                     // tile_dim1=1, tile_dim2=0
  g1[5] = (int)td0;                              // tensor_dim0_stride lo
  g1[6] = 0;                                     // stride hi / dim1_stride lo
  g1[7] = 0;
  i32x4 z4 = {0, 0, 0, 0};
#if __has_include(<hip/amd_detail/amd_gfx1250_TDM.h>)
  i32x8 z8 = {0, 0, 0, 0, 0, 0, 0, 0};
  __builtin_amdgcn_tensor_load_to_lds(g0, g1, z4, z4, z8, 0);   // clang-23 6-arg
#else
  __builtin_amdgcn_tensor_load_to_lds(g0, g1, z4, z4, 0);       // ROCm7.2 5-arg
#endif
}
#endif

// ---------------------------------------------------------------------------
// WMMA helpers (wave32, V_WMMA_F32_16X16X32_F16)
// A fragment: lane = M row (lane&15); vgprs 0-3 hold K 0..7 (lanes<16) or
// 8..15 (lanes>=16); vgprs 4-7 hold K 16..23 / 24..31.  Row-major source
// => two contiguous 8-half chunks per lane.
// B fragment: identical gather pattern applied to W^T (N x K row-major).
// ---------------------------------------------------------------------------
__device__ __forceinline__ v16h load_frag16(const _Float16* __restrict__ p, int ld) {
  const int lane = threadIdx.x & 31;
  const int r    = lane & 15;
  const int kb   = (lane & 16) ? 8 : 0;
  const _Float16* q = p + r * ld + kb;
  v16h out;
#pragma unroll
  for (int i = 0; i < 8; ++i) out[i] = q[i];
#pragma unroll
  for (int i = 0; i < 8; ++i) out[8 + i] = q[16 + i];
  return out;
}

__device__ __forceinline__ v8f wmma16(v16h a, v16h b, v8f c) {
  return __builtin_amdgcn_wmma_f32_16x16x32_f16(false, a, false, b, (short)0, c,
                                                false, false);
}

// One MLP layer inside a workgroup: Y = relu(X @ W^T' + b), X: CHUNK x KD (LDS,
// tight), W^T: (CTN*16) x KD (global f16), Y: CHUNK x (CTN*16) (LDS, tight).
template <int RT, int CTN, int KD>
__device__ __forceinline__ void mlp_layer(const _Float16* Xs, _Float16* Ys,
                                          const _Float16* __restrict__ wt,
                                          const float* __restrict__ bias,
                                          int wave, int lane) {
  const int LDO = CTN * 16;
  for (int t = wave; t < RT * CTN; t += 8) {
    const int rt = t / CTN, ct = t % CTN;
    v8f c = {};
#pragma unroll
    for (int kt = 0; kt < KD / 32; ++kt) {
      v16h a = load_frag16(Xs + (rt * 16) * KD + kt * 32, KD);
      v16h b = load_frag16(wt + (ct * 16) * KD + kt * 32, KD);
      c = wmma16(a, b, c);
    }
    const int n  = ct * 16 + (lane & 15);
    const float bb = bias[n];
    const int mb = rt * 16 + ((lane & 16) ? 8 : 0);
#pragma unroll
    for (int rr = 0; rr < 8; ++rr) {
      float v = c[rr] + bb;
      v = v > 0.f ? v : 0.f;
      Ys[(mb + rr) * LDO + n] = (_Float16)v;
    }
  }
}

// ---------------------------------------------------------------------------
// Fused Set-Abstraction layer: one workgroup per center; stream CHUNK source
// points through 3-layer MLP in LDS, masked max into registers.
// ---------------------------------------------------------------------------
template <int FIN, int L0, int L1, int L2, int CHUNK, bool USE_MASK>
__global__ void __launch_bounds__(256)
sa_kernel(const float* __restrict__ srcC,        // N x 3
          const _Float16* __restrict__ featpad,  // N x FIN (cols 0..2 scratch)
          const float* __restrict__ centers,     // K x 3
          const _Float16* __restrict__ wt0, const float* __restrict__ b0,
          const _Float16* __restrict__ wt1, const float* __restrict__ b1,
          const _Float16* __restrict__ wt2, const float* __restrict__ b2,
          float r2, int N, float* __restrict__ out)  // K x L2
{
  constexpr int RT = CHUNK / 16, CT0 = L0 / 16, CT1 = L1 / 16;
  constexpr int CT2 = L2 / 16, CTW = CT2 / 8;
  __shared__ _Float16 sX[CHUNK * FIN];
  __shared__ _Float16 sY0[CHUNK * L0];
  __shared__ _Float16 sY1[CHUNK * L1];
  __shared__ float    sOob[CHUNK];

  const int tid = threadIdx.x, wave = tid >> 5, lane = tid & 31;
  const int cidx = blockIdx.x;
  const float cx = centers[cidx * 3 + 0];
  const float cy = centers[cidx * 3 + 1];
  const float cz = centers[cidx * 3 + 2];

  float acc[CTW][8];
#pragma unroll
  for (int a = 0; a < CTW; ++a)
#pragma unroll
    for (int b = 0; b < 8; ++b) acc[a][b] = -__builtin_inff();

  for (int base = 0; base < N; base += CHUNK) {
    // ---- stage padded feature chunk into LDS ----
#ifdef USE_TDM
    if (tid < 32) {  // wave 0 drives the Tensor Data Mover
      tdm_copy_to_lds(featpad + (size_t)base * FIN, sX, CHUNK * FIN);
      __builtin_amdgcn_s_wait_tensorcnt(0);
    }
#else
    {
      const unsigned* srcU = (const unsigned*)(featpad + (size_t)base * FIN);
      unsigned* dstU = (unsigned*)sX;
      for (int i = tid; i < CHUNK * FIN / 2; i += 256) dstU[i] = srcU[i];
    }
#endif
    if (base + CHUNK < N)  // prefetch next chunk into cache hierarchy
      __builtin_prefetch(featpad + (size_t)(base + CHUNK) * FIN, 0, 3);
    __syncthreads();
    // overwrite cols 0..2 with relative coords; compute radius mask
    for (int i = tid; i < CHUNK; i += 256) {
      const int g = base + i;
      float dx = srcC[g * 3 + 0] - cx;
      float dy = srcC[g * 3 + 1] - cy;
      float dz = srcC[g * 3 + 2] - cz;
      sX[i * FIN + 0] = (_Float16)dx;
      sX[i * FIN + 1] = (_Float16)dy;
      sX[i * FIN + 2] = (_Float16)dz;
      float d2 = dx * dx + dy * dy + dz * dz;
      sOob[i] = (d2 < r2) ? 0.f : 1.f;
    }
    __syncthreads();

    mlp_layer<RT, CT0, FIN>(sX, sY0, wt0, b0, wave, lane);
    __syncthreads();
    mlp_layer<RT, CT1, L0>(sY0, sY1, wt1, b1, wave, lane);
    __syncthreads();

    // layer 2 + masked running max; each wave owns CTW column tiles
#pragma unroll
    for (int cw = 0; cw < CTW; ++cw) {
      const int ct = wave * CTW + cw;
      const int n  = ct * 16 + (lane & 15);
      const float bb = b2[n];
      for (int rt = 0; rt < RT; ++rt) {
        v8f c = {};
#pragma unroll
        for (int kt = 0; kt < L1 / 32; ++kt) {
          v16h a = load_frag16(sY1 + (rt * 16) * L1 + kt * 32, L1);
          v16h b = load_frag16(wt2 + (ct * 16) * L1 + kt * 32, L1);
          c = wmma16(a, b, c);
        }
        const int mb = rt * 16 + ((lane & 16) ? 8 : 0);
#pragma unroll
        for (int rr = 0; rr < 8; ++rr) {
          float v = c[rr] + bb;
          v = v > 0.f ? v : 0.f;
          if (USE_MASK && sOob[mb + rr] != 0.f) v = -__builtin_inff();
          acc[cw][rr] = fmaxf(acc[cw][rr], v);
        }
      }
    }
    __syncthreads();
  }

  // reduce: 8 vgpr rows within lane, then rows 0-7 vs 8-15 across half-waves
#pragma unroll
  for (int cw = 0; cw < CTW; ++cw) {
    float m = acc[cw][0];
#pragma unroll
    for (int rr = 1; rr < 8; ++rr) m = fmaxf(m, acc[cw][rr]);
    m = fmaxf(m, __shfl_xor(m, 16, 32));
    if (lane < 16) {
      const int ct = wave * CTW + cw;
      out[(size_t)cidx * L2 + ct * 16 + lane] = m;
    }
  }
}

// ---------------------------------------------------------------------------
// Generic WMMA GEMM: C = A(M x Kpad, f16) @ W^T(Npad x Kpad, f16)^T + b, fp32 out
// ---------------------------------------------------------------------------
__global__ void __launch_bounds__(256)
gemm_bias_kernel(const _Float16* __restrict__ A, int Kpad,
                 const _Float16* __restrict__ WT,
                 const float* __restrict__ bias,
                 float* __restrict__ C, int M, int Nreal, int Npad, int relu) {
  const int wg = blockIdx.x * 8 + (threadIdx.x >> 5);
  const int CT = Npad / 16, RT = M / 16;
  if (wg >= RT * CT) return;
  const int rt = wg / CT, ct = wg % CT;
  v8f c = {};
  for (int kt = 0; kt < Kpad / 32; ++kt) {
    v16h a = load_frag16(A + (size_t)(rt * 16) * Kpad + kt * 32, Kpad);
    v16h b = load_frag16(WT + (size_t)(ct * 16) * Kpad + kt * 32, Kpad);
    c = wmma16(a, b, c);
  }
  const int lane = threadIdx.x & 31;
  const int n  = ct * 16 + (lane & 15);
  const int hi = (lane & 16) ? 8 : 0;
  const float bb = (n < Nreal) ? bias[n] : 0.f;
#pragma unroll
  for (int rr = 0; rr < 8; ++rr) {
    float v = c[rr] + bb;
    if (relu) v = fmaxf(v, 0.f);
    if (n < Nreal) C[(size_t)(rt * 16 + rr + hi) * Nreal + n] = v;
  }
}

// ---------------------------------------------------------------------------
// Farthest point sampling: single workgroup, sequential scan
// ---------------------------------------------------------------------------
__global__ void __launch_bounds__(256)
fps_kernel(const float* __restrict__ coords, int N, int K, int* __restrict__ idx) {
  __shared__ float smind[2048];
  __shared__ float svals[256];
  __shared__ int   sidx[256];
  __shared__ int   s_last;
  const int tid = threadIdx.x;
  for (int i = tid; i < N; i += 256) smind[i] = 3.4e38f;
  if (tid == 0) { idx[0] = 0; s_last = 0; }
  __syncthreads();
  for (int it = 1; it < K; ++it) {
    const int last = s_last;
    const float lx = coords[last * 3], ly = coords[last * 3 + 1],
                lz = coords[last * 3 + 2];
    float best = -1.f;
    int bi = 0;
    for (int i = tid; i < N; i += 256) {
      float dx = coords[i * 3] - lx, dy = coords[i * 3 + 1] - ly,
            dz = coords[i * 3 + 2] - lz;
      float d = dx * dx + dy * dy + dz * dz;
      float mnd = fminf(smind[i], d);
      smind[i] = mnd;
      if (mnd > best) { best = mnd; bi = i; }
    }
    svals[tid] = best; sidx[tid] = bi;
    __syncthreads();
    for (int s = 128; s > 0; s >>= 1) {
      if (tid < s) {
        float v2 = svals[tid + s]; int i2 = sidx[tid + s];
        if (v2 > svals[tid] || (v2 == svals[tid] && i2 < sidx[tid])) {
          svals[tid] = v2; sidx[tid] = i2;
        }
      }
      __syncthreads();
    }
    if (tid == 0) { idx[it] = sidx[0]; s_last = sidx[0]; }
    __syncthreads();
  }
}

__global__ void gather3_kernel(const float* __restrict__ src,
                               const int* __restrict__ idx, int K,
                               float* __restrict__ dst) {
  int i = blockIdx.x * blockDim.x + threadIdx.x;
  if (i >= K * 3) return;
  int k = i / 3, c = i % 3;
  dst[i] = src[idx[k] * 3 + c];
}

// ---------------------------------------------------------------------------
// 3-NN inverse-square-distance interpolation (p = 2)
// ---------------------------------------------------------------------------
__global__ void knn3_kernel(const float* __restrict__ toC, int NT,
                            const float* __restrict__ fromC, int NF,
                            int* __restrict__ oidx, float* __restrict__ ow) {
  int t = blockIdx.x * blockDim.x + threadIdx.x;
  if (t >= NT) return;
  const float tx = toC[t * 3], ty = toC[t * 3 + 1], tz = toC[t * 3 + 2];
  float d0 = 3.4e38f, d1 = 3.4e38f, d2 = 3.4e38f;
  int i0 = 0, i1 = 0, i2 = 0;
  for (int j = 0; j < NF; ++j) {
    float dx = tx - fromC[j * 3], dy = ty - fromC[j * 3 + 1],
          dz = tz - fromC[j * 3 + 2];
    float d = dx * dx + dy * dy + dz * dz;
    if (d < d0)      { d2 = d1; i2 = i1; d1 = d0; i1 = i0; d0 = d; i0 = j; }
    else if (d < d1) { d2 = d1; i2 = i1; d1 = d; i1 = j; }
    else if (d < d2) { d2 = d; i2 = j; }
  }
  const int k = NF < 3 ? NF : 3;
  float dd[3] = {d0, d1, d2};
  int   ii[3] = {i0, i1, i2};
  bool anyz = false;
  for (int j = 0; j < k; ++j) anyz |= (dd[j] == 0.f);
  float wr[3] = {0.f, 0.f, 0.f};
  if (anyz) {
    for (int j = 0; j < k; ++j) wr[j] = (dd[j] == 0.f) ? 1.f : 0.f;
  } else {
    float s = 0.f;
    for (int j = 0; j < k; ++j) { wr[j] = 1.f / dd[j]; s += wr[j]; }
    for (int j = 0; j < k; ++j) wr[j] /= s;
  }
  for (int j = 0; j < 3; ++j) {
    oidx[t * 3 + j] = (j < k) ? ii[j] : ii[k - 1];
    ow[t * 3 + j]   = (j < k) ? wr[j] : 0.f;
  }
}

__global__ void interp_gather_kernel(const float* __restrict__ feats, int F,
                                     const int* __restrict__ idx,
                                     const float* __restrict__ w, int NT,
                                     float* __restrict__ out) {
  int i = blockIdx.x * blockDim.x + threadIdx.x;
  if (i >= NT * F) return;
  int t = i / F, f = i % F;
  float acc = 0.f;
#pragma unroll
  for (int j = 0; j < 3; ++j)
    acc += w[t * 3 + j] * feats[(size_t)idx[t * 3 + j] * F + f];
  out[i] = acc;
}

__global__ void broadcast_row_kernel(const float* __restrict__ row, int F,
                                     int M, float* __restrict__ out) {
  int i = blockIdx.x * blockDim.x + threadIdx.x;
  if (i >= M * F) return;
  out[i] = row[i % F];
}

// ---------------------------------------------------------------------------
// BatchNorm (over rows) + pack helpers
// ---------------------------------------------------------------------------
__global__ void bn_stats_kernel(const float* __restrict__ X, int M, int N,
                                float* __restrict__ mean, float* __restrict__ var) {
  int n = blockIdx.x * blockDim.x + threadIdx.x;
  if (n >= N) return;
  float s = 0.f, s2 = 0.f;
  for (int m = 0; m < M; ++m) {
    float v = X[(size_t)m * N + n];
    s += v; s2 += v * v;
  }
  float mu = s / M;
  mean[n] = mu;
  var[n] = s2 / M - mu * mu;
}

__global__ void bn_apply_kernel(const float* __restrict__ X, int M, int N,
                                const float* __restrict__ mean,
                                const float* __restrict__ var,
                                const float* __restrict__ g,
                                const float* __restrict__ bt,
                                _Float16* dst16, int ld16, int coff,
                                float* dst32) {
  int i = blockIdx.x * blockDim.x + threadIdx.x;
  if (i >= M * N) return;
  int m = i / N, n = i % N;
  float v = (X[i] - mean[n]) * rsqrtf(var[n] + BN_EPS) * g[n] + bt[n];
  v = fmaxf(v, 0.f);
  if (dst16) dst16[(size_t)m * ld16 + coff + n] = (_Float16)v;
  if (dst32) dst32[i] = v;
}

// Transpose+pad fp32 weight (fin x fout) -> f16 W^T (npad x kpad)
__global__ void pack_wt_kernel(const float* __restrict__ wsrc,
                               _Float16* __restrict__ wt, int fin, int fout,
                               int kpad, int npad) {
  int i = blockIdx.x * blockDim.x + threadIdx.x;
  if (i >= npad * kpad) return;
  int n = i / kpad, k = i % kpad;
  wt[i] = (_Float16)((n < fout && k < fin) ? wsrc[(size_t)k * fout + n] : 0.f);
}

// Copy fp32 (M x F) into f16 padded matrix (ld columns) at column offset
__global__ void pack_cols_kernel(const float* __restrict__ src, int M, int F,
                                 _Float16* __restrict__ dst, int ld, int coff) {
  int i = blockIdx.x * blockDim.x + threadIdx.x;
  if (i >= M * F) return;
  int m = i / F, f = i % F;
  dst[(size_t)m * ld + coff + f] = (_Float16)src[i];
}

// ---------------------------------------------------------------------------
// Host orchestration
// ---------------------------------------------------------------------------
#define CDIV(a, b) (((a) + (b) - 1) / (b))

extern "C" void kernel_launch(void* const* d_in, const int* in_sizes, int n_in,
                              void* d_out, int out_size, void* d_ws,
                              size_t ws_size, hipStream_t stream) {
  (void)in_sizes; (void)n_in; (void)out_size; (void)ws_size;
  // tree_leaves order (dict keys sorted): coords, features, then params:
  // end(b,w x3), fp0(b,bt,g,w x2), fp1(... x2), fp2(l0:b,bt,g,w; l1:b,w),
  // sa1(b,w x3), sa2(b,w x3)
  const float* coords   = (const float*)d_in[0];
  const float* features = (const float*)d_in[1];
  const float* end_b0 = (const float*)d_in[2];  const float* end_w0 = (const float*)d_in[3];
  const float* end_b1 = (const float*)d_in[4];  const float* end_w1 = (const float*)d_in[5];
  const float* end_b2 = (const float*)d_in[6];  const float* end_w2 = (const float*)d_in[7];
  const float* fp0_b0 = (const float*)d_in[8];  const float* fp0_bt0 = (const float*)d_in[9];
  const float* fp0_g0 = (const float*)d_in[10]; const float* fp0_w0  = (const float*)d_in[11];
  const float* fp0_b1 = (const float*)d_in[12]; const float* fp0_bt1 = (const float*)d_in[13];
  const float* fp0_g1 = (const float*)d_in[14]; const float* fp0_w1  = (const float*)d_in[15];
  const float* fp1_b0 = (const float*)d_in[16]; const float* fp1_bt0 = (const float*)d_in[17];
  const float* fp1_g0 = (const float*)d_in[18]; const float* fp1_w0  = (const float*)d_in[19];
  const float* fp1_b1 = (const float*)d_in[20]; const float* fp1_bt1 = (const float*)d_in[21];
  const float* fp1_g1 = (const float*)d_in[22]; const float* fp1_w1  = (const float*)d_in[23];
  const float* fp2_b0 = (const float*)d_in[24]; const float* fp2_bt0 = (const float*)d_in[25];
  const float* fp2_g0 = (const float*)d_in[26]; const float* fp2_w0  = (const float*)d_in[27];
  const float* fp2_b1 = (const float*)d_in[28]; const float* fp2_w1  = (const float*)d_in[29];
  const float* sa1_b0 = (const float*)d_in[30]; const float* sa1_w0 = (const float*)d_in[31];
  const float* sa1_b1 = (const float*)d_in[32]; const float* sa1_w1 = (const float*)d_in[33];
  const float* sa1_b2 = (const float*)d_in[34]; const float* sa1_w2 = (const float*)d_in[35];
  const float* sa2_b0 = (const float*)d_in[36]; const float* sa2_w0 = (const float*)d_in[37];
  const float* sa2_b1 = (const float*)d_in[38]; const float* sa2_w1 = (const float*)d_in[39];
  const float* sa2_b2 = (const float*)d_in[40]; const float* sa2_w2 = (const float*)d_in[41];

  // ---- workspace bump allocator ----
  char* base = (char*)d_ws;
  size_t off = 0;
  auto alloc = [&](size_t bytes) -> void* {
    void* p = base + off;
    off += (bytes + 255) & ~(size_t)255;
    return p;
  };
  // f16 transposed/padded weights
  _Float16* wt_sa1_0 = (_Float16*)alloc(64 * 32 * 2);
  _Float16* wt_sa1_1 = (_Float16*)alloc(64 * 64 * 2);
  _Float16* wt_sa1_2 = (_Float16*)alloc(128 * 64 * 2);
  _Float16* wt_sa2_0 = (_Float16*)alloc(128 * 160 * 2);
  _Float16* wt_sa2_1 = (_Float16*)alloc(128 * 128 * 2);
  _Float16* wt_sa2_2 = (_Float16*)alloc(256 * 128 * 2);
  _Float16* wt_end_0 = (_Float16*)alloc(256 * 288 * 2);
  _Float16* wt_end_1 = (_Float16*)alloc(512 * 256 * 2);
  _Float16* wt_end_2 = (_Float16*)alloc((size_t)1024 * 512 * 2);
  _Float16* wt_fp0_0 = (_Float16*)alloc((size_t)256 * 1280 * 2);
  _Float16* wt_fp0_1 = (_Float16*)alloc(256 * 256 * 2);
  _Float16* wt_fp1_0 = (_Float16*)alloc(256 * 384 * 2);
  _Float16* wt_fp1_1 = (_Float16*)alloc(128 * 256 * 2);
  _Float16* wt_fp2_0 = (_Float16*)alloc(128 * 160 * 2);
  _Float16* wt_fp2_1 = (_Float16*)alloc(64 * 128 * 2);
  // f16 activation/pads
  _Float16* featpad1 = (_Float16*)alloc((size_t)2048 * 32 * 2);
  _Float16* featpad2 = (_Float16*)alloc((size_t)512 * 160 * 2);
  _Float16* featpad3 = (_Float16*)alloc((size_t)128 * 288 * 2);
  _Float16* fp0in    = (_Float16*)alloc((size_t)128 * 1280 * 2);
  _Float16* fp1in    = (_Float16*)alloc((size_t)512 * 384 * 2);
  _Float16* fp2in    = (_Float16*)alloc((size_t)2048 * 160 * 2);
  _Float16* h16a     = (_Float16*)alloc((size_t)128 * 256 * 2);
  _Float16* h16b     = (_Float16*)alloc((size_t)512 * 256 * 2);
  _Float16* h16c     = (_Float16*)alloc((size_t)2048 * 128 * 2);
  // fp32
  float* c1      = (float*)alloc(512 * 3 * 4);
  float* c2      = (float*)alloc(128 * 3 * 4);
  float* f1      = (float*)alloc((size_t)512 * 128 * 4);
  float* f2      = (float*)alloc((size_t)128 * 256 * 4);
  float* f3      = (float*)alloc(1024 * 4);
  float* interp0 = (float*)alloc((size_t)128 * 1024 * 4);
  float* f2p     = (float*)alloc((size_t)128 * 256 * 4);
  float* interp1 = (float*)alloc((size_t)512 * 256 * 4);
  float* f1p     = (float*)alloc((size_t)512 * 128 * 4);
  float* interp2 = (float*)alloc((size_t)2048 * 128 * 4);
  float* zbuf    = (float*)alloc((size_t)2048 * 128 * 4);  // largest BN input
  float* meanb   = (float*)alloc(256 * 4);
  float* varb    = (float*)alloc(256 * 4);
  int*   idx1    = (int*)alloc(512 * 4);
  int*   idx2    = (int*)alloc(128 * 4);
  int*   kidx    = (int*)alloc((size_t)2048 * 3 * 4);
  float* kw      = (float*)alloc((size_t)2048 * 3 * 4);

  // ---- weight prep ----
  auto packW = [&](const float* w, _Float16* wt, int fin, int fout, int kpad,
                   int npad) {
    int n = npad * kpad;
    pack_wt_kernel<<<CDIV(n, 256), 256, 0, stream>>>(w, wt, fin, fout, kpad, npad);
  };
  packW(sa1_w0, wt_sa1_0, 9, 64, 32, 64);
  packW(sa1_w1, wt_sa1_1, 64, 64, 64, 64);
  packW(sa1_w2, wt_sa1_2, 64, 128, 64, 128);
  packW(sa2_w0, wt_sa2_0, 131, 128, 160, 128);
  packW(sa2_w1, wt_sa2_1, 128, 128, 128, 128);
  packW(sa2_w2, wt_sa2_2, 128, 256, 128, 256);
  packW(end_w0, wt_end_0, 259, 256, 288, 256);
  packW(end_w1, wt_end_1, 256, 512, 256, 512);
  packW(end_w2, wt_end_2, 512, 1024, 512, 1024);
  packW(fp0_w0, wt_fp0_0, 1280, 256, 1280, 256);
  packW(fp0_w1, wt_fp0_1, 256, 256, 256, 256);
  packW(fp1_w0, wt_fp1_0, 384, 256, 384, 256);
  packW(fp1_w1, wt_fp1_1, 256, 128, 256, 128);
  packW(fp2_w0, wt_fp2_0, 134, 128, 160, 128);
  packW(fp2_w1, wt_fp2_1, 128, 50, 128, 64);

  auto gemm = [&](const _Float16* A, int Kpad, const _Float16* WT,
                  const float* bias, float* C, int M, int Nreal, int Npad) {
    int tiles = (M / 16) * (Npad / 16);
    gemm_bias_kernel<<<CDIV(tiles, 8), 256, 0, stream>>>(A, Kpad, WT, bias, C,
                                                         M, Nreal, Npad, 0);
  };

  // ---- SA1 ----
  hipMemsetAsync(featpad1, 0, (size_t)2048 * 32 * 2, stream);
  pack_cols_kernel<<<CDIV(2048 * 6, 256), 256, 0, stream>>>(features, 2048, 6,
                                                            featpad1, 32, 3);
  fps_kernel<<<1, 256, 0, stream>>>(coords, 2048, 512, idx1);
  gather3_kernel<<<CDIV(512 * 3, 256), 256, 0, stream>>>(coords, idx1, 512, c1);
  sa_kernel<32, 64, 64, 128, 128, true><<<512, 256, 0, stream>>>(
      coords, featpad1, c1, wt_sa1_0, sa1_b0, wt_sa1_1, sa1_b1, wt_sa1_2,
      sa1_b2, 0.2f * 0.2f, 2048, f1);

  // ---- SA2 ----
  hipMemsetAsync(featpad2, 0, (size_t)512 * 160 * 2, stream);
  pack_cols_kernel<<<CDIV(512 * 128, 256), 256, 0, stream>>>(f1, 512, 128,
                                                             featpad2, 160, 3);
  fps_kernel<<<1, 256, 0, stream>>>(c1, 512, 128, idx2);
  gather3_kernel<<<CDIV(128 * 3, 256), 256, 0, stream>>>(c1, idx2, 128, c2);
  sa_kernel<160, 128, 128, 256, 64, true><<<128, 256, 0, stream>>>(
      c1, featpad2, c2, wt_sa2_0, sa2_b0, wt_sa2_1, sa2_b1, wt_sa2_2, sa2_b2,
      0.4f * 0.4f, 512, f2);

  // ---- SA_end (single center = c2[0], no mask) ----
  hipMemsetAsync(featpad3, 0, (size_t)128 * 288 * 2, stream);
  pack_cols_kernel<<<CDIV(128 * 256, 256), 256, 0, stream>>>(f2, 128, 256,
                                                             featpad3, 288, 3);
  sa_kernel<288, 256, 512, 1024, 16, false><<<1, 256, 0, stream>>>(
      c2, featpad3, c2, wt_end_0, end_b0, wt_end_1, end_b1, wt_end_2, end_b2,
      1e30f, 128, f3);

  // ---- FP0: concat(f2, interp(f3)) -> 256 -> 256 ----
  broadcast_row_kernel<<<CDIV(128 * 1024, 256), 256, 0, stream>>>(f3, 1024, 128,
                                                                  interp0);
  hipMemsetAsync(fp0in, 0, (size_t)128 * 1280 * 2, stream);
  pack_cols_kernel<<<CDIV(128 * 256, 256), 256, 0, stream>>>(f2, 128, 256,
                                                             fp0in, 1280, 0);
  pack_cols_kernel<<<CDIV(128 * 1024, 256), 256, 0, stream>>>(
      interp0, 128, 1024, fp0in, 1280, 256);
  gemm(fp0in, 1280, wt_fp0_0, fp0_b0, zbuf, 128, 256, 256);
  bn_stats_kernel<<<1, 256, 0, stream>>>(zbuf, 128, 256, meanb, varb);
  bn_apply_kernel<<<CDIV(128 * 256, 256), 256, 0, stream>>>(
      zbuf, 128, 256, meanb, varb, fp0_g0, fp0_bt0, h16a, 256, 0, nullptr);
  gemm(h16a, 256, wt_fp0_1, fp0_b1, zbuf, 128, 256, 256);
  bn_stats_kernel<<<1, 256, 0, stream>>>(zbuf, 128, 256, meanb, varb);
  bn_apply_kernel<<<CDIV(128 * 256, 256), 256, 0, stream>>>(
      zbuf, 128, 256, meanb, varb, fp0_g1, fp0_bt1, nullptr, 0, 0, f2p);

  // ---- FP1: concat(f1, interp(f2p: c2->c1)) -> 256 -> 128 ----
  knn3_kernel<<<CDIV(512, 256), 256, 0, stream>>>(c1, 512, c2, 128, kidx, kw);
  interp_gather_kernel<<<CDIV(512 * 256, 256), 256, 0, stream>>>(
      f2p, 256, kidx, kw, 512, interp1);
  hipMemsetAsync(fp1in, 0, (size_t)512 * 384 * 2, stream);
  pack_cols_kernel<<<CDIV(512 * 128, 256), 256, 0, stream>>>(f1, 512, 128,
                                                             fp1in, 384, 0);
  pack_cols_kernel<<<CDIV(512 * 256, 256), 256, 0, stream>>>(
      interp1, 512, 256, fp1in, 384, 128);
  gemm(fp1in, 384, wt_fp1_0, fp1_b0, zbuf, 512, 256, 256);
  bn_stats_kernel<<<1, 256, 0, stream>>>(zbuf, 512, 256, meanb, varb);
  bn_apply_kernel<<<CDIV(512 * 256, 256), 256, 0, stream>>>(
      zbuf, 512, 256, meanb, varb, fp1_g0, fp1_bt0, h16b, 256, 0, nullptr);
  gemm(h16b, 256, wt_fp1_1, fp1_b1, zbuf, 512, 128, 128);
  bn_stats_kernel<<<1, 256, 0, stream>>>(zbuf, 512, 128, meanb, varb);
  bn_apply_kernel<<<CDIV(512 * 128, 256), 256, 0, stream>>>(
      zbuf, 512, 128, meanb, varb, fp1_g1, fp1_bt1, nullptr, 0, 0, f1p);

  // ---- FP2: concat(features, interp(f1p: c1->coords)) -> 128 -> 50 ----
  knn3_kernel<<<CDIV(2048, 256), 256, 0, stream>>>(coords, 2048, c1, 512, kidx, kw);
  interp_gather_kernel<<<CDIV(2048 * 128, 256), 256, 0, stream>>>(
      f1p, 128, kidx, kw, 2048, interp2);
  hipMemsetAsync(fp2in, 0, (size_t)2048 * 160 * 2, stream);
  pack_cols_kernel<<<CDIV(2048 * 6, 256), 256, 0, stream>>>(features, 2048, 6,
                                                            fp2in, 160, 0);
  pack_cols_kernel<<<CDIV(2048 * 128, 256), 256, 0, stream>>>(
      interp2, 2048, 128, fp2in, 160, 6);
  gemm(fp2in, 160, wt_fp2_0, fp2_b0, zbuf, 2048, 128, 128);
  bn_stats_kernel<<<1, 256, 0, stream>>>(zbuf, 2048, 128, meanb, varb);
  bn_apply_kernel<<<CDIV(2048 * 128, 256), 256, 0, stream>>>(
      zbuf, 2048, 128, meanb, varb, fp2_g0, fp2_bt0, h16c, 128, 0, nullptr);
  // final layer: bias only, no BN/relu, fp32 out (2048 x 50)
  gemm(h16c, 128, wt_fp2_1, fp2_b1, (float*)d_out, 2048, 50, 64);
}
